// SlidingWindowAttention_10428180594723
// MI455X (gfx1250) — matrix operand order repaired and checked by reading, not compiled, not run on hardware
//
#include <hip/hip_runtime.h>
#include <hip/hip_bf16.h>

// ---------------------------------------------------------------------------
// CDNA5 (gfx1250) sliding-window attention, f16 WMMA everywhere.
// Wave32; V_WMMA_F32_16X16X32_F16 fragment layouts per CDNA5 ISA 7.12.2:
//   A (16x32 f16):  row = lane&15, half i -> K = (i&7) + (i>>3)*16 + (lane>>4)*8
//   B (32x16 f16):  col = lane&15, same K order as A; this code loads A and B as K = i + (lane>>4)*16
//   C/D (16x16 f32): vgpr j -> M = j + 8*(lane>>4), N = lane&15
// GEMM staging uses GLOBAL_LOAD_ASYNC_TO_LDS_B128 (ASYNCcnt) per ISA ch.8 §4.
// ---------------------------------------------------------------------------

#ifndef USE_ASYNC_STAGE
#define USE_ASYNC_STAGE 1   // flip to 0 to fall back to load+ds_store staging
#endif

typedef __attribute__((ext_vector_type(16))) _Float16 v16h;
typedef __attribute__((ext_vector_type(8)))  float    v8f;

#define N_HEADS 16
#define D_HEAD  64
#define WINDOW  256
#define TSEQ    2048
#define BATCH   2
#define DMODEL  1024

__device__ __forceinline__ v8f wmma16(v16h a, v16h b, v8f c) {
  return __builtin_amdgcn_wmma_f32_16x16x32_f16(
      /*neg_a=*/false, a, /*neg_b=*/false, b,
      /*c_mod=*/(short)0, c, /*reuse_a=*/false, /*reuse_b=*/false);
}

// B fragment: 16 contiguous halves starting at p (lane's column-run)
__device__ __forceinline__ v16h load16_f16(const _Float16* __restrict__ p) {
  v16h b;
#pragma unroll
  for (int i = 0; i < 16; ++i) b[i] = p[i];
  return b;
}
// A fragment lane-run: halves 0..15 at p[0..15]
__device__ __forceinline__ v16h loadA_f16(const _Float16* __restrict__ p) {
  v16h a;
#pragma unroll
  for (int i = 0; i < 16; ++i) a[i] = p[i];
  return a;
}

#if USE_ASYNC_STAGE
// Async 16B global -> LDS copy. Generic LDS pointers are {aperture, off[31:0]},
// so the low 32 bits are the raw LDS byte address the instruction wants.
__device__ __forceinline__ void async_b128(void* lds, const void* g) {
  const unsigned soff = (unsigned)(size_t)lds;
  const unsigned long long ga = (unsigned long long)(size_t)g;
  asm volatile("global_load_async_to_lds_b128 %0, %1, off"
               :: "v"(soff), "v"(ga) : "memory");
}
__device__ __forceinline__ void wait_async0() {
  asm volatile("s_wait_asynccnt 0" ::: "memory");
}
#endif

// ---------------------------------------------------------------------------
// Kernel 0: one-shot f32 -> f16 conversion (memory bound, ~6us total).
// ---------------------------------------------------------------------------
__global__ __launch_bounds__(256) void swa_cvt_kernel(
    const float* __restrict__ src, _Float16* __restrict__ dst) {
  const int i = (blockIdx.x * 256 + threadIdx.x) * 4;
  const float4 v = *(const float4*)(src + i);
  dst[i + 0] = (_Float16)v.x;
  dst[i + 1] = (_Float16)v.y;
  dst[i + 2] = (_Float16)v.z;
  dst[i + 3] = (_Float16)v.w;
}

// ---------------------------------------------------------------------------
// Unified f16 GEMM: Y[m,n] = A[m,:] . Bm[n,:], K = 1024.
// Block = 8 waves = 128x128 tile (wave tile 32x64, waves arranged 4 x 2).
// A/B slabs (128x32 f16 = 8KB each) staged in LDS, double-buffered, filled by
// async-to-LDS copies overlapped with the WMMAs of the current slab.
// mode 0: M=4096, N=4096 (QKV+gate projection epilogue, scatter + sigmoid)
// mode 1: M=4096, N=1024 (output projection, f32 store)
// ---------------------------------------------------------------------------
__global__ __launch_bounds__(256) void swa_gemm_kernel(
    const _Float16* __restrict__ A, const _Float16* __restrict__ Bm, int mode,
    _Float16* __restrict__ qk_ws,   // [2][B][H][T][Dh] f16 (part 0=Q, 1=K)
    _Float16* __restrict__ vt_ws,   // [B][H][Dh][T]    f16 (V transposed)
    _Float16* __restrict__ gate_ws, // [B][T][1024]     f16 (sigmoid applied)
    float* __restrict__ y)          // [B][T][1024]     f32 (mode 1)
{
  __shared__ _Float16 sA[2][128 * 32];
  __shared__ _Float16 sB[2][128 * 32];

  const int tid  = threadIdx.x;
  const int lane = tid & 31;
  const int wave = tid >> 5;
  const int bm   = (blockIdx.x & 31) * 128;
  const int bn   = (blockIdx.x >> 5) * 128;
  const int wm   = (wave & 3) * 32;    // wave M offset within block tile
  const int wn   = (wave >> 2) * 64;   // wave N offset within block tile
  const int nl   = lane & 15;
  const int gh   = (lane >> 4);

  v8f acc[2][4] = {};

  // Stage one 32-k slab of A and B (each thread moves 2 x 16B per matrix).
  auto stage = [&](int buf, int kb) {
#pragma unroll
    for (int c = 0; c < 2; ++c) {
      const int idx = c * 256 + tid;       // 512 chunks of 16B per matrix
      const int row = idx >> 2;
      const int col = (idx & 3) * 8;
#if USE_ASYNC_STAGE
      async_b128(sA[buf] + row * 32 + col,
                 A + (size_t)(bm + row) * DMODEL + kb + col);
      async_b128(sB[buf] + row * 32 + col,
                 Bm + (size_t)(bn + row) * DMODEL + kb + col);
#else
      *(float4*)(sA[buf] + row * 32 + col) =
          *(const float4*)(A + (size_t)(bm + row) * DMODEL + kb + col);
      *(float4*)(sB[buf] + row * 32 + col) =
          *(const float4*)(Bm + (size_t)(bn + row) * DMODEL + kb + col);
#endif
    }
  };

  stage(0, 0);
#if USE_ASYNC_STAGE
  wait_async0();
#endif
  __syncthreads();

  int buf = 0;
  for (int kb = 0; kb < DMODEL; kb += 32) {
    if (kb + 32 < DMODEL) stage(buf ^ 1, kb + 32);   // overlaps with WMMAs

    v16h af[2];
#pragma unroll
    for (int sm = 0; sm < 2; ++sm)
      af[sm] = loadA_f16(sA[buf] + (wm + sm * 16 + nl) * 32 + gh * 16);
    v16h bf[4];
#pragma unroll
    for (int sn = 0; sn < 4; ++sn)
      bf[sn] = load16_f16(sB[buf] + (wn + sn * 16 + nl) * 32 + gh * 16);

#pragma unroll
    for (int sm = 0; sm < 2; ++sm)
#pragma unroll
      for (int sn = 0; sn < 4; ++sn)
        acc[sm][sn] = wmma16(af[sm], bf[sn], acc[sm][sn]);

#if USE_ASYNC_STAGE
    wait_async0();      // next slab landed (in-order ASYNCcnt), this wave clean
#endif
    __syncthreads();    // publish next slab; all reads of `buf` are done
    buf ^= 1;
  }

  // ---- epilogue ----
  if (mode == 1) {
#pragma unroll
    for (int sm = 0; sm < 2; ++sm)
#pragma unroll
      for (int sn = 0; sn < 4; ++sn)
#pragma unroll
        for (int j = 0; j < 8; ++j) {
          const int m = bm + wm + sm * 16 + j + gh * 8;
          const int n = bn + wn + sn * 16 + nl;
          y[(size_t)m * DMODEL + n] = acc[sm][sn][j];
        }
    return;
  }

#pragma unroll
  for (int sm = 0; sm < 2; ++sm) {
#pragma unroll
    for (int sn = 0; sn < 4; ++sn) {
#pragma unroll
      for (int j = 0; j < 8; ++j) {
        const int m = bm + wm + sm * 16 + j + gh * 8;
        const int n = bn + wn + sn * 16 + nl;
        const float val = acc[sm][sn][j];
        const int b = m >> 11;           // T = 2048
        const int t = m & (TSEQ - 1);
        if (n < 2048) {                  // Q (part 0) or K (part 1)
          const int part = n >> 10;
          const int h = (n & 1023) >> 6;
          const int d = n & 63;
          qk_ws[((((size_t)part * BATCH + b) * N_HEADS + h) * TSEQ + t) * D_HEAD + d] =
              (_Float16)val;
        } else if (n < 3072) {           // V transposed: [B][H][Dh][T]
          const int h = (n & 1023) >> 6;
          const int d = n & 63;
          vt_ws[(((size_t)b * N_HEADS + h) * D_HEAD + d) * TSEQ + t] = (_Float16)val;
        } else {                         // gate, sigmoid
          gate_ws[(size_t)m * DMODEL + (n & 1023)] =
              (_Float16)(1.0f / (1.0f + __expf(-val)));
        }
      }
    }
  }
}

// ---------------------------------------------------------------------------
// Kernel 2: sliding-window flash attention + gating.
// One wave per (b, h, 16-query block): 2*16*128 = 4096 waves -> 512 blocks.
// 32-key blocks; P staged through per-wave LDS for the cross-lane transpose
// into A-fragment layout (wave-synchronous via s_wait_dscnt).
// ---------------------------------------------------------------------------
__global__ __launch_bounds__(256) void swa_attn_kernel(
    const _Float16* __restrict__ qk_ws, const _Float16* __restrict__ vt_ws,
    const _Float16* __restrict__ gate_ws, _Float16* __restrict__ ag_ws)
{
  __shared__ _Float16 plds[8][16 * 32];   // per-wave P staging, 1 KiB each

  const int lane = threadIdx.x & 31;
  const int wave = threadIdx.x >> 5;
  const int wid  = blockIdx.x * 8 + wave;
  const int q0   = (wid & 127) * 16;
  const int h    = (wid >> 7) & (N_HEADS - 1);
  const int b    = wid >> 11;
  const int nl   = lane & 15;
  const int hi   = (lane >> 4) * 8;   // C-fragment row offset
  const int gh   = (lane >> 4);

  const _Float16* qbase = qk_ws + (((size_t)0 * BATCH + b) * N_HEADS + h) * TSEQ * D_HEAD;
  const _Float16* kbase = qk_ws + (((size_t)1 * BATCH + b) * N_HEADS + h) * TSEQ * D_HEAD;
  const _Float16* vtb   = vt_ws + (((size_t)b * N_HEADS + h) * D_HEAD) * TSEQ;

  // Q fragments: rows q0..q0+15, K-dim = Dh split into [0,32) and [32,64)
  const _Float16* qp = qbase + (size_t)(q0 + nl) * D_HEAD + gh * 16;
  const v16h qa0 = loadA_f16(qp);
  const v16h qa1 = loadA_f16(qp + 32);

  v8f o[4] = {};
  float rmax[8], rsum[8];
#pragma unroll
  for (int j = 0; j < 8; ++j) { rmax[j] = -1e30f; rsum[j] = 0.0f; }

  const float scale = 0.125f;                            // Dh^-1/2
  const int klo = (q0 > 255) ? ((q0 - 255) & ~31) : 0;   // 32-aligned

  for (int kb = klo; kb <= q0 + 15; kb += 32) {
    // ---- S = Q . K^T for 32 keys (two 16x16 C tiles) ----
    v8f s0 = {}, s1 = {};
    {
      const _Float16* k0 = kbase + (size_t)(kb + nl) * D_HEAD + gh * 16;
      const _Float16* k1 = kbase + (size_t)(kb + 16 + nl) * D_HEAD + gh * 16;
      s0 = wmma16(qa0, load16_f16(k0), s0);
      s0 = wmma16(qa1, load16_f16(k0 + 32), s0);
      s1 = wmma16(qa0, load16_f16(k1), s1);
      s1 = wmma16(qa1, load16_f16(k1 + 32), s1);
    }

    // ---- mask + online softmax (row stats live per half-wave group) ----
    float p0[8], p1[8], alpha[8];
#pragma unroll
    for (int j = 0; j < 8; ++j) {
      const int q    = q0 + j + hi;
      const int key0 = kb + nl;
      const int key1 = kb + 16 + nl;
      const int d0 = q - key0, d1 = q - key1;
      const bool v0 = (d0 >= 0) && (d0 < WINDOW);
      const bool v1 = (d1 >= 0) && (d1 < WINDOW);
      const float s0v = v0 ? s0[j] * scale : -1e30f;
      const float s1v = v1 ? s1[j] * scale : -1e30f;

      float lmax = fmaxf(s0v, s1v);
#pragma unroll
      for (int off = 1; off < 16; off <<= 1)
        lmax = fmaxf(lmax, __shfl_xor(lmax, off, 32));  // stays in 16-lane half

      const float nm = fmaxf(rmax[j], lmax);
      const float a  = __expf(rmax[j] - nm);
      const float e0 = v0 ? __expf(s0v - nm) : 0.0f;
      const float e1 = v1 ? __expf(s1v - nm) : 0.0f;
      float ls = e0 + e1;
#pragma unroll
      for (int off = 1; off < 16; off <<= 1)
        ls += __shfl_xor(ls, off, 32);

      rsum[j] = rsum[j] * a + ls;
      rmax[j] = nm;
      alpha[j] = a;
      p0[j] = e0; p1[j] = e1;
    }

    // ---- stage P (16x32) through LDS: C layout -> A-fragment layout ----
    _Float16* pl = plds[wave];
#pragma unroll
    for (int j = 0; j < 8; ++j) {
      const int r = j + hi;
      pl[r * 32 + nl]      = (_Float16)p0[j];
      pl[r * 32 + 16 + nl] = (_Float16)p1[j];
    }
    asm volatile("s_wait_dscnt 0" ::: "memory");     // wave-synchronous LDS
    const v16h pf = loadA_f16(pl + (size_t)nl * 32 + gh * 16);

    // ---- O = O*alpha + P . V  (4 d-tiles of 16) ----
#pragma unroll
    for (int t = 0; t < 4; ++t) {
#pragma unroll
      for (int j = 0; j < 8; ++j) o[t][j] *= alpha[j];
      const int d = t * 16 + nl;
      const _Float16* vp = vtb + (size_t)d * TSEQ + kb + gh * 16;
      o[t] = wmma16(pf, load16_f16(vp), o[t]);
    }
  }

  // ---- normalize, gate, store f16 A-operand for the output projection ----
  float inv[8];
#pragma unroll
  for (int j = 0; j < 8; ++j) inv[j] = 1.0f / rsum[j];

  const _Float16* gp = gate_ws + (size_t)b * TSEQ * DMODEL;
  _Float16*       ap = ag_ws   + (size_t)b * TSEQ * DMODEL;
#pragma unroll
  for (int t = 0; t < 4; ++t) {
#pragma unroll
    for (int j = 0; j < 8; ++j) {
      const int q   = q0 + j + hi;
      const int col = h * D_HEAD + t * 16 + nl;
      const float val = o[t][j] * inv[j];
      const float g   = (float)gp[(size_t)q * DMODEL + col];
      ap[(size_t)q * DMODEL + col] = (_Float16)(val * g);
    }
  }
}

// ---------------------------------------------------------------------------
// Host launcher. Workspace layout (58 MiB total):
//   [ 0,16M) qk_ws f16 [2][B][H][T][Dh]
//   [16,24M) vt_ws f16 [B][H][Dh][T]
//   [24,32M) gate  f16 [B][T][1024]
//   [32,40M) ag    f16 [B][T][1024]
//   [40,48M) xh    f16 [B*T][1024]
//   [48,56M) wh    f16 [4096][1024]  (w_qkv rows 0..3071, w_gate rows 3072..4095)
//   [56,58M) wouth f16 [1024][1024]
// ---------------------------------------------------------------------------
extern "C" void kernel_launch(void* const* d_in, const int* in_sizes, int n_in,
                              void* d_out, int out_size, void* d_ws, size_t ws_size,
                              hipStream_t stream) {
  const float* x      = (const float*)d_in[0];
  const float* w_qkv  = (const float*)d_in[1];
  const float* w_gate = (const float*)d_in[2];
  const float* w_out  = (const float*)d_in[3];
  float* out = (float*)d_out;

  char* ws = (char*)d_ws;
  const size_t MB = 1024 * 1024;
  _Float16* qk    = (_Float16*)(ws);
  _Float16* vt    = (_Float16*)(ws + 16 * MB);
  _Float16* gate  = (_Float16*)(ws + 24 * MB);
  _Float16* ag    = (_Float16*)(ws + 32 * MB);
  _Float16* xh    = (_Float16*)(ws + 40 * MB);
  _Float16* wh    = (_Float16*)(ws + 48 * MB);
  _Float16* wouth = (_Float16*)(ws + 56 * MB);

  // f32 -> f16 pre-conversion (1024 elements per block).
  swa_cvt_kernel<<<4096, 256, 0, stream>>>(x, xh);                 // 4.19M
  swa_cvt_kernel<<<3072, 256, 0, stream>>>(w_qkv, wh);             // 3.15M
  swa_cvt_kernel<<<1024, 256, 0, stream>>>(w_gate, wh + (size_t)3072 * DMODEL);
  swa_cvt_kernel<<<1024, 256, 0, stream>>>(w_out, wouth);

  // QKV + gate projection: M=4096, N=4096 -> 32x32 = 1024 blocks.
  swa_gemm_kernel<<<1024, 256, 0, stream>>>(xh, wh, 0, qk, vt, gate, nullptr);
  // Attention: 4096 waves -> 512 blocks.
  swa_attn_kernel<<<512, 256, 0, stream>>>(qk, vt, gate, ag);
  // Output projection: M=4096, N=1024 -> 32x8 = 256 blocks.
  swa_gemm_kernel<<<256, 256, 0, stream>>>(ag, wouth, 1, nullptr, nullptr, nullptr, out);
}
